// GroupedQueryAttention_35029753266734
// MI455X (gfx1250) — matrix operand order repaired and checked by reading, not compiled, not run on hardware
//
#include <hip/hip_runtime.h>

// ---------------------------------------------------------------------------
// GQA attention for MI455X (gfx1250), bf16 WMMA pipeline.
// B=2, S=2048, E=2048, H=32, KV=8, D=64, G=4, scale = 1/8.
// ---------------------------------------------------------------------------

typedef __bf16 bf16_t;
typedef __attribute__((ext_vector_type(8)))  __bf16 v8bf;
typedef __attribute__((ext_vector_type(16))) __bf16 v16bf;
typedef __attribute__((ext_vector_type(8)))  float  v8f;

#define CB 2
#define CS 2048
#define CE 2048
#define CH 32
#define CKV 8
#define CD 64
#define CSCALE 0.125f

static __device__ __forceinline__ v8bf ld8(const bf16_t* p) {
    return *reinterpret_cast<const v8bf*>(p);
}

static __device__ __forceinline__ v16bf concat8(v8bf lo, v8bf hi) {
    return __builtin_shufflevector(lo, hi, 0,1,2,3,4,5,6,7,8,9,10,11,12,13,14,15);
}

static __device__ __forceinline__ v8f wmma_bf16(v16bf a, v16bf b, v8f c) {
    return __builtin_amdgcn_wmma_f32_16x16x32_bf16(
        /*neg_a=*/false, a, /*neg_b=*/false, b,
        /*c_mod=*/(short)0, c, /*reuse_a=*/false, /*reuse_b=*/false);
}

static __device__ __forceinline__ v8f vzero8() {
    v8f z;
#pragma unroll
    for (int i = 0; i < 8; ++i) z[i] = 0.0f;
    return z;
}

// ---------------------------------------------------------------------------
// Tiled GEMM: C[M,N] = A[M,K] * W[K,N].  A: TA (fp32 or bf16), W: fp32,
// C: TO (bf16 or fp32).  Block = 128 threads (4 waves), 64x64 output tile,
// K-step = 64 (8 WMMAs per wave per LDS staging round).
// ---------------------------------------------------------------------------
template <typename TA, typename TO>
__global__ void gemm64_wmma(const TA* __restrict__ A, const float* __restrict__ W,
                            TO* __restrict__ C, int M, int N, int K) {
    __shared__ __align__(16) bf16_t As[64][64];   // [m][k], k contiguous
    __shared__ __align__(16) bf16_t WT[64][64];   // [n][k], k contiguous (transposed)

    const int tid  = threadIdx.x;
    const int wave = tid >> 5;
    const int lane = tid & 31;
    const int half = lane >> 4;
    const int ln   = lane & 15;
    const int m0 = blockIdx.y * 64;
    const int n0 = blockIdx.x * 64;

    v8f acc[4];
#pragma unroll
    for (int t = 0; t < 4; ++t) acc[t] = vzero8();

    for (int k0 = 0; k0 < K; k0 += 64) {
        // Stage A tile (64x64) with fp32->bf16 conversion.
#pragma unroll
        for (int i = tid; i < 64 * 64; i += 128) {
            int mm = i >> 6, kk = i & 63;
            As[mm][kk] = (bf16_t)(float)A[(size_t)(m0 + mm) * K + (k0 + kk)];
        }
        // Stage W tile (64x64) transposed -> WT[n][k].
#pragma unroll
        for (int i = tid; i < 64 * 64; i += 128) {
            int kk = i >> 6, nn = i & 63;
            WT[nn][kk] = (bf16_t)W[(size_t)(k0 + kk) * N + (n0 + nn)];
        }
        __syncthreads();

        const int row = wave * 16 + ln;
#pragma unroll
        for (int c = 0; c < 2; ++c) {
            // A fragment: lane row = wave*16 + ln; K split {half*8..+7, 16+half*8..+7}
            v16bf afrag = concat8(ld8(&As[row][c * 32 + half * 8]),
                                  ld8(&As[row][c * 32 + 16 + half * 8]));
#pragma unroll
            for (int t = 0; t < 4; ++t) {
                // B fragment: lane col = t*16 + ln; K split {half*8..+7, 16+half*8..+7}
                const int col = t * 16 + ln;
                v16bf bfrag = concat8(ld8(&WT[col][c * 32 + half * 8]),
                                      ld8(&WT[col][c * 32 + 16 + half * 8]));
                acc[t] = wmma_bf16(afrag, bfrag, acc[t]);
            }
        }
        __syncthreads();
    }

    // C/D layout: element (m, n) -> vgpr r = m - 8*half, lane n (per half).
#pragma unroll
    for (int t = 0; t < 4; ++t) {
#pragma unroll
        for (int r = 0; r < 8; ++r) {
            const int m = m0 + wave * 16 + 8 * half + r;
            const int n = n0 + t * 16 + ln;
            C[(size_t)m * N + n] = (TO)acc[t][r];
        }
    }
}

// ---------------------------------------------------------------------------
// RoPE (interleaved) + head transpose: in[B][S][NH*64] -> out[B][NH][S][64].
// One thread per even/odd pair.  Optionally no-rope (V) and scale (Q: 1/sqrt(D)).
// ---------------------------------------------------------------------------
__global__ void rope_pack(const bf16_t* __restrict__ in, bf16_t* __restrict__ out,
                          const float* __restrict__ cosT, const float* __restrict__ sinT,
                          int NH, int doRope, float scale) {
    const int idx = blockIdx.x * blockDim.x + threadIdx.x;
    const int total = CB * CS * NH * (CD / 2);
    if (idx >= total) return;
    const int d2 = idx & 31;                 // pair index within head
    const int h  = (idx >> 5) % NH;
    const int s  = (idx / (32 * NH)) % CS;
    const int b  = idx / (32 * NH * CS);
    const int d0 = d2 * 2;

    const size_t ibase = (((size_t)b * CS + s) * NH + h) * CD + d0;
    float x0 = (float)in[ibase];
    float x1 = (float)in[ibase + 1];
    float y0 = x0, y1 = x1;
    if (doRope) {
        const float c  = cosT[s * CD + d0];  // repeated per pair
        const float sn = sinT[s * CD + d0];
        y0 = x0 * c - x1 * sn;
        y1 = x1 * c + x0 * sn;
    }
    y0 *= scale; y1 *= scale;
    const size_t obase = (((size_t)b * NH + h) * CS + s) * CD + d0;
    out[obase]     = (bf16_t)y0;
    out[obase + 1] = (bf16_t)y1;
}

// ---------------------------------------------------------------------------
// Flash attention (causal, GQA).  Grid: (S/64, H, B), block = 128 (4 waves).
// Each wave owns 16 query rows with private online-softmax state; K/V tiles
// (32 keys) staged in LDS cooperatively.  QK^T and P*V both via bf16 WMMA.
// Q layout [B][H][S][D], K/V layout [B][KV][S][D], all bf16.  Q pre-scaled.
// The per-wave causal bound is scalarized (readfirstlane) so the skip branch
// is an s_cbranch and EXEC stays all-ones around every WMMA (ISA 7.12 req).
// ---------------------------------------------------------------------------
__global__ void attn_flash_wmma(const bf16_t* __restrict__ Q,
                                const bf16_t* __restrict__ Kt,
                                const bf16_t* __restrict__ Vt,
                                bf16_t* __restrict__ O) {
    __shared__ __align__(16) bf16_t Ks[32][64];      // [key][d]
    __shared__ __align__(16) bf16_t VsT[64][32];     // [d][key] (transposed)
    __shared__ __align__(16) bf16_t Ps[4][16][32];   // per-wave P staging [q][key]

    const int tid  = threadIdx.x;
    const int wave = tid >> 5;
    const int lane = tid & 31;
    const int half = lane >> 4;
    const int ln   = lane & 15;

    const int h   = blockIdx.y;
    const int b   = blockIdx.z;
    const int kvh = h >> 2;                          // G = 4
    const int qbase = (blockIdx.x * 4 + wave) * 16;  // wave's 16 query rows

    // Scalar (SGPR) copy of this wave's causal key-block bound.
    const int mykb = __builtin_amdgcn_readfirstlane((qbase + 15) / 32 + 1);

    // Preload both Q A-fragments (D=64 -> two K=32 steps) straight from global:
    // row-major [q][d] matches A-fragment contiguity requirements.
    const bf16_t* Qrow = Q + ((((size_t)b * CH + h) * CS) + qbase + ln) * CD;
    const v16bf qf0 = concat8(ld8(Qrow + half * 8),      ld8(Qrow + 16 + half * 8));
    const v16bf qf1 = concat8(ld8(Qrow + 32 + half * 8), ld8(Qrow + 48 + half * 8));

    float mrow[8], lrow[8];
    v8f oacc[4];
#pragma unroll
    for (int r = 0; r < 8; ++r) { mrow[r] = -1e30f; lrow[r] = 0.0f; }
#pragma unroll
    for (int t = 0; t < 4; ++t) oacc[t] = vzero8();

    const bf16_t* Kbase = Kt + (((size_t)b * CKV + kvh) * CS) * CD;
    const bf16_t* Vbase = Vt + (((size_t)b * CKV + kvh) * CS) * CD;

    const int nkb = blockIdx.x * 2 + 2;  // key blocks of 32 up to block diag
    for (int kb = 0; kb < nkb; ++kb) {
        const int kbase = kb * 32;

        // Cooperative K tile copy: 32x64 bf16 = 256 uint4.
        const uint4* ksrc = (const uint4*)(Kbase + (size_t)kbase * CD);
        uint4* kdst = (uint4*)(&Ks[0][0]);
        kdst[tid]       = ksrc[tid];
        kdst[tid + 128] = ksrc[tid + 128];

        // Cooperative V tile copy with transpose into VsT[d][key].
        const uint4* vsrc = (const uint4*)(Vbase + (size_t)kbase * CD);
#pragma unroll
        for (int i = 0; i < 2; ++i) {
            const int idx = tid + i * 128;
            const int key = idx >> 3, dc = idx & 7;
            union { uint4 u; bf16_t e[8]; } tmp;
            tmp.u = vsrc[idx];
#pragma unroll
            for (int j = 0; j < 8; ++j) VsT[dc * 8 + j][key] = tmp.e[j];
        }

        // Warm L2/WGP$ with the next tiles while this one computes
        // (lowers to global_prefetch_b8; no counters, fire-and-forget).
        if (kb + 1 < nkb) {
            __builtin_prefetch(Kbase + (size_t)(kbase + 32) * CD, 0, 0);
            __builtin_prefetch(Vbase + (size_t)(kbase + 32) * CD, 0, 0);
        }
        __syncthreads();

        if (kb < mykb) {  // scalar branch: EXEC untouched, all-ones for WMMA
            // ---- scores S = Q K^T (16 q x 32 keys), two K-dim steps ----
            v8f s0 = vzero8(), s1 = vzero8();
            {
                v16bf b0 = concat8(ld8(&Ks[ln][half * 8]),      ld8(&Ks[ln][16 + half * 8]));
                v16bf b1 = concat8(ld8(&Ks[16 + ln][half * 8]), ld8(&Ks[16 + ln][16 + half * 8]));
                s0 = wmma_bf16(qf0, b0, s0);
                s1 = wmma_bf16(qf0, b1, s1);
            }
            {
                v16bf b0 = concat8(ld8(&Ks[ln][32 + half * 8]),      ld8(&Ks[ln][48 + half * 8]));
                v16bf b1 = concat8(ld8(&Ks[16 + ln][32 + half * 8]), ld8(&Ks[16 + ln][48 + half * 8]));
                s0 = wmma_bf16(qf1, b0, s0);
                s1 = wmma_bf16(qf1, b1, s1);
            }

            // ---- causal mask (only diagonal-crossing blocks) ----
            if (kbase + 31 > qbase) {
#pragma unroll
                for (int r = 0; r < 8; ++r) {
                    const int qr = qbase + 8 * half + r;
                    if (kbase + ln > qr)      s0[r] = -1e30f;
                    if (kbase + 16 + ln > qr) s1[r] = -1e30f;
                }
            }

            // ---- online softmax: per-row max/sum across 16-lane half ----
#pragma unroll
            for (int r = 0; r < 8; ++r) {
                float mx = fmaxf(s0[r], s1[r]);
                mx = fmaxf(mx, __shfl_xor(mx, 8, 32));
                mx = fmaxf(mx, __shfl_xor(mx, 4, 32));
                mx = fmaxf(mx, __shfl_xor(mx, 2, 32));
                mx = fmaxf(mx, __shfl_xor(mx, 1, 32));
                const float mnew  = fmaxf(mrow[r], mx);
                const float p0    = __expf(s0[r] - mnew);
                const float p1    = __expf(s1[r] - mnew);
                const float alpha = __expf(mrow[r] - mnew);
                float rs = p0 + p1;
                rs += __shfl_xor(rs, 8, 32);
                rs += __shfl_xor(rs, 4, 32);
                rs += __shfl_xor(rs, 2, 32);
                rs += __shfl_xor(rs, 1, 32);
                lrow[r] = lrow[r] * alpha + rs;
                mrow[r] = mnew;
                s0[r] = p0;
                s1[r] = p1;
                oacc[0][r] *= alpha; oacc[1][r] *= alpha;
                oacc[2][r] *= alpha; oacc[3][r] *= alpha;
            }

            // ---- restage P from C-layout to A-fragment layout via LDS ----
#pragma unroll
            for (int r = 0; r < 8; ++r) {
                Ps[wave][8 * half + r][ln]      = (bf16_t)s0[r];
                Ps[wave][8 * half + r][16 + ln] = (bf16_t)s1[r];
            }
            asm volatile("s_wait_dscnt 0" ::: "memory");  // same-wave RAW fence
            v16bf pf = concat8(ld8(&Ps[wave][ln][half * 8]),
                               ld8(&Ps[wave][ln][16 + half * 8]));

            // ---- O += P * V (K=32 keys, four 16-wide d tiles) ----
#pragma unroll
            for (int t = 0; t < 4; ++t) {
                v16bf bv = concat8(ld8(&VsT[t * 16 + ln][half * 8]),
                                   ld8(&VsT[t * 16 + ln][16 + half * 8]));
                oacc[t] = wmma_bf16(pf, bv, oacc[t]);
            }
        }
        __syncthreads();
    }

    // ---- normalize and store O[b][q][h*64 + d] (bf16) ----
#pragma unroll
    for (int r = 0; r < 8; ++r) {
        const float inv = 1.0f / lrow[r];
        const int q = qbase + 8 * half + r;
        const size_t orow = ((size_t)b * CS + q) * (CH * CD) + h * CD;
#pragma unroll
        for (int t = 0; t < 4; ++t)
            O[orow + t * 16 + ln] = (bf16_t)(oacc[t][r] * inv);
    }
}

// ---------------------------------------------------------------------------
// Launch: 3 projection GEMMs -> RoPE/pack x3 -> flash attention -> Wo GEMM.
// ---------------------------------------------------------------------------
extern "C" void kernel_launch(void* const* d_in, const int* in_sizes, int n_in,
                              void* d_out, int out_size, void* d_ws, size_t ws_size,
                              hipStream_t stream) {
    const float* hidden = (const float*)d_in[0];
    const float* cosT   = (const float*)d_in[1];
    const float* sinT   = (const float*)d_in[2];
    const float* Wq     = (const float*)d_in[3];
    const float* Wk     = (const float*)d_in[4];
    const float* Wv     = (const float*)d_in[5];
    const float* Wo     = (const float*)d_in[6];
    float* out = (float*)d_out;

    // Workspace carve-up (bytes).  Total = 48 MiB.
    char* ws = (char*)d_ws;
    const size_t SZ_QPROJ = (size_t)CB * CS * CH * CD * 2;    // 16 MiB
    const size_t SZ_KPROJ = (size_t)CB * CS * CKV * CD * 2;   // 4 MiB
    bf16_t* qproj = (bf16_t*)(ws);
    bf16_t* kproj = (bf16_t*)(ws + SZ_QPROJ);
    bf16_t* vproj = (bf16_t*)(ws + SZ_QPROJ + SZ_KPROJ);
    bf16_t* Qp    = (bf16_t*)(ws + SZ_QPROJ + 2 * SZ_KPROJ);
    bf16_t* Kp    = (bf16_t*)(ws + 2 * SZ_QPROJ + 2 * SZ_KPROJ);
    bf16_t* Vp    = (bf16_t*)(ws + 2 * SZ_QPROJ + 3 * SZ_KPROJ);
    bf16_t* Obuf  = qproj;  // qproj is dead after rope_pack(Q); reuse for attn out

    const dim3 blk(128);
    const int M = CB * CS;  // 4096

    // QKV projections (fp32 in, bf16 out), bf16 WMMA.
    gemm64_wmma<float, bf16_t><<<dim3((CH * CD) / 64, M / 64), blk, 0, stream>>>(
        hidden, Wq, qproj, M, CH * CD, CE);
    gemm64_wmma<float, bf16_t><<<dim3((CKV * CD) / 64, M / 64), blk, 0, stream>>>(
        hidden, Wk, kproj, M, CKV * CD, CE);
    gemm64_wmma<float, bf16_t><<<dim3((CKV * CD) / 64, M / 64), blk, 0, stream>>>(
        hidden, Wv, vproj, M, CKV * CD, CE);

    // RoPE + transpose to [B][heads][S][D]; Q also gets the 1/sqrt(D) scale.
    const int tq = CB * CS * CH * (CD / 2);
    const int tk = CB * CS * CKV * (CD / 2);
    rope_pack<<<(tq + 255) / 256, 256, 0, stream>>>(qproj, Qp, cosT, sinT, CH, 1, CSCALE);
    rope_pack<<<(tk + 255) / 256, 256, 0, stream>>>(kproj, Kp, cosT, sinT, CKV, 1, 1.0f);
    rope_pack<<<(tk + 255) / 256, 256, 0, stream>>>(vproj, Vp, cosT, sinT, CKV, 0, 1.0f);

    // Causal flash attention.
    attn_flash_wmma<<<dim3(CS / 64, CH, CB), blk, 0, stream>>>(Qp, Kp, Vp, Obuf);

    // Output projection (bf16 in, fp32 out).
    gemm64_wmma<bf16_t, float><<<dim3(CE / 64, M / 64), blk, 0, stream>>>(
        Obuf, Wo, out, M, CE, CH * CD);
}
